// RNNClassifier_19799799234901
// MI455X (gfx1250) — compile-verified, matrix operand order checked by reading, not run on hardware
//
#include <hip/hip_runtime.h>
#include <hip/hip_bf16.h>

// ---------------- problem constants ----------------
constexpr int kB = 128;
constexpr int kT = 500;
constexpr int kF = 64;
constexpr int kU = 512;
constexpr int kNC = 11;
constexpr int k4U = 4 * kU;          // 2048
constexpr int kNT = k4U / 16;        // 128 column tiles of 16
constexpr int kK0 = kF + kU;         // 576  = cell0 K
constexpr int kK1 = 2 * kU;          // 1024 = cell1 K
constexpr int kKT0 = kK0 / 32;       // 18 bf16 WMMA K-steps
constexpr int kKT1 = kK1 / 32;       // 32

// ---------------- CDNA5 WMMA types ----------------
typedef __attribute__((ext_vector_type(16))) __bf16 v16bf;
typedef __attribute__((ext_vector_type(8)))  float  v8f;

union FragU {
    uint4 q[2];
    v16bf v;
};

__device__ inline unsigned short f32_to_bf16(float f) {
    unsigned u = __float_as_uint(f);
    u += 0x7FFFu + ((u >> 16) & 1u);           // round-to-nearest-even
    return (unsigned short)(u >> 16);
}

__device__ inline float sigmoidf_(float x) { return 1.0f / (1.0f + __expf(-x)); }

__device__ inline float time_gate(float t, float tau, float sh) {
    float x = (t - sh) / tau;
    float phi = x - floorf(x);                  // jnp.mod semantics, in [0,1)
    float k;
    if (phi < 0.025f)      k = phi * 40.0f;            // 2*phi/R_ON
    else if (phi < 0.05f)  k = 2.0f - phi * 40.0f;     // 2 - 2*phi/R_ON
    else                   k = 0.001f * phi;           // ALPHA*phi
    return k;
}

// ---------------- prologue: zero recurrent state ----------------
__global__ void zero_state_kernel(unsigned* p, int n) {
    int i = blockIdx.x * blockDim.x + threadIdx.x;
    for (; i < n; i += gridDim.x * blockDim.x) p[i] = 0u;
}

// ---------------- prologue: pack weights into WMMA-B tile order -------------
// Packed layout: tile(kt, nt) = 32 lanes x 16 bf16, contiguous per lane (32B).
// Lane l (<16): col = nt*16+l, k-offsets {0..7, 16..23}; lane l (>=16):
// col = nt*16+(l-16), k-offsets {8..15, 24..31}.  (ISA 05_wmma B layout.)
__global__ void pack_weights_kernel(const float* __restrict__ k0,
                                    const float* __restrict__ rk0,
                                    const float* __restrict__ k1,
                                    const float* __restrict__ rk1,
                                    unsigned short* __restrict__ w0p,
                                    unsigned short* __restrict__ w1p) {
    const int n0 = kKT0 * kNT * 32 * 16;       // cell0 packed elements
    const int n1 = kKT1 * kNT * 32 * 16;       // cell1 packed elements
    for (int i = blockIdx.x * blockDim.x + threadIdx.x; i < n0 + n1;
         i += gridDim.x * blockDim.x) {
        const bool second = (i >= n0);
        const int j = second ? (i - n0) : i;
        const int e    = j & 15;
        const int lane = (j >> 4) & 31;
        const int nt   = (j >> 9) & (kNT - 1);
        const int kt   = j >> 16;              // 128*32*16 = 65536 per k-step
        const int kb   = (lane & 16) ? 8 : 0;
        const int k    = kt * 32 + kb + ((e < 8) ? e : e + 8);
        const int n    = nt * 16 + (lane & 15);
        float v;
        if (second) v = (k < kU) ? k1[k * k4U + n] : rk1[(k - kU) * k4U + n];
        else        v = (k < kF) ? k0[k * k4U + n] : rk0[(k - kF) * k4U + n];
        (second ? w1p : w0p)[j] = f32_to_bf16(v);
    }
}

// ---------------- PhasedLSTM cell 0 (fused GEMM + gates + time gate) --------
// grid (8,8): x = batch tile (16 rows), y = unit tile (64 units).
// 4 waves; wave g owns gate g -> 4 accumulator tiles, K = 576.
__global__ __launch_bounds__(128) void plstm_cell0(
    const float* __restrict__ xin,            // [B,T,F]
    const float* __restrict__ times,          // [B,T]
    const unsigned short* __restrict__ w0p,   // packed bf16 [kKT0][kNT][32][16]
    const float* __restrict__ bias,           // [4U]
    const float* __restrict__ tau,            // [U]
    const float* __restrict__ shift,          // [U]
    const float* __restrict__ hprevf,         // [B,U] fp32
    const unsigned short* __restrict__ hprevb,// [B,U] bf16
    float* __restrict__ cst,                  // [B,U] fp32 (in place)
    float* __restrict__ hnewf,                // [B,U] fp32
    unsigned short* __restrict__ hnewb,       // [B,U] bf16
    int tstep) {
    __shared__ unsigned short A_s[16 * kK0];  // 18 KB
    __shared__ float z_s[4][16][64];          // 16 KB

    const int tid  = threadIdx.x;
    const int wave = tid >> 5;
    const int lane = tid & 31;
    const int bm0  = blockIdx.x * 16;
    const int u0   = blockIdx.y * 64;

    // Stage A = [x_t | h_prev] (bf16) into LDS, shared by all 4 gate-waves.
    for (int idx = tid; idx < 16 * kK0; idx += 128) {
        const int m = idx / kK0;
        const int k = idx - m * kK0;
        unsigned short v;
        if (k < kF) v = f32_to_bf16(xin[((bm0 + m) * kT + tstep) * kF + k]);
        else        v = hprevb[(bm0 + m) * kU + (k - kF)];
        A_s[idx] = v;
    }
    __syncthreads();

    v8f acc[4] = {};
    const int nt0  = wave * (kU / 16) + (u0 >> 4);   // first column tile
    const int kb   = (lane & 16) ? 8 : 0;
    const int mrow = lane & 15;

    for (int kt = 0; kt < kKT0; ++kt) {
        FragU a;
        const int abase = mrow * kK0 + kt * 32 + kb;
        a.q[0] = *(const uint4*)&A_s[abase];
        a.q[1] = *(const uint4*)&A_s[abase + 16];
#pragma unroll
        for (int nt = 0; nt < 4; ++nt) {
            FragU bfr;
            const uint4* bp =
                (const uint4*)&w0p[((kt * kNT + (nt0 + nt)) * 32 + lane) * 16];
            bfr.q[0] = bp[0];
            bfr.q[1] = bp[1];
            acc[nt] = __builtin_amdgcn_wmma_f32_16x16x32_bf16(
                false, a.v, false, bfr.v, (short)0, acc[nt], false, false);
        }
    }

    // Scatter D tiles to LDS (C/D layout: lane<16 -> M=r, lane>=16 -> M=r+8).
#pragma unroll
    for (int nt = 0; nt < 4; ++nt)
#pragma unroll
        for (int r = 0; r < 8; ++r) {
            const int row = r + ((lane & 16) ? 8 : 0);
            z_s[wave][row][nt * 16 + (lane & 15)] = acc[nt][r];
        }
    __syncthreads();

    // Elementwise gates + phased time gate.
    for (int idx = tid; idx < 16 * 64; idx += 128) {
        const int row = idx >> 6, col = idx & 63;
        const int b = bm0 + row, u = u0 + col;
        const float zi = z_s[0][row][col] + bias[0 * kU + u];
        const float zf = z_s[1][row][col] + bias[1 * kU + u];
        const float zg = z_s[2][row][col] + bias[2 * kU + u];
        const float zo = z_s[3][row][col] + bias[3 * kU + u];
        const float gi = sigmoidf_(zi), gf = sigmoidf_(zf), go = sigmoidf_(zo);
        const float gg = tanhf(zg);
        const float cp = cst[b * kU + u];
        const float ct = gf * cp + gi * gg;
        const float ht = go * tanhf(ct);
        const float kt = time_gate(times[b * kT + tstep], tau[u], shift[u]);
        const float cn = kt * ct + (1.0f - kt) * cp;
        const float hn = kt * ht + (1.0f - kt) * hprevf[b * kU + u];
        cst[b * kU + u]   = cn;
        hnewf[b * kU + u] = hn;
        hnewb[b * kU + u] = f32_to_bf16(hn);
    }
}

// ---------------- PhasedLSTM cell 1 (A = [h0_new | h1_prev], K = 1024) ------
__global__ __launch_bounds__(128) void plstm_cell1(
    const float* __restrict__ times,
    const unsigned short* __restrict__ w1p,
    const float* __restrict__ bias,
    const float* __restrict__ tau,
    const float* __restrict__ shift,
    const unsigned short* __restrict__ h0curb,  // cell0 output bf16
    const float* __restrict__ hprevf,
    const unsigned short* __restrict__ hprevb,
    float* __restrict__ cst,
    float* __restrict__ hnewf,
    unsigned short* __restrict__ hnewb,
    int tstep) {
    __shared__ unsigned short A_s[16 * kK1];  // 32 KB
    __shared__ float z_s[4][16][64];          // 16 KB

    const int tid  = threadIdx.x;
    const int wave = tid >> 5;
    const int lane = tid & 31;
    const int bm0  = blockIdx.x * 16;
    const int u0   = blockIdx.y * 64;

    for (int idx = tid; idx < 16 * kK1; idx += 128) {
        const int m = idx >> 10;          // / 1024
        const int k = idx & (kK1 - 1);
        A_s[idx] = (k < kU) ? h0curb[(bm0 + m) * kU + k]
                            : hprevb[(bm0 + m) * kU + (k - kU)];
    }
    __syncthreads();

    v8f acc[4] = {};
    const int nt0  = wave * (kU / 16) + (u0 >> 4);
    const int kb   = (lane & 16) ? 8 : 0;
    const int mrow = lane & 15;

    for (int kt = 0; kt < kKT1; ++kt) {
        FragU a;
        const int abase = mrow * kK1 + kt * 32 + kb;
        a.q[0] = *(const uint4*)&A_s[abase];
        a.q[1] = *(const uint4*)&A_s[abase + 16];
#pragma unroll
        for (int nt = 0; nt < 4; ++nt) {
            FragU bfr;
            const uint4* bp =
                (const uint4*)&w1p[((kt * kNT + (nt0 + nt)) * 32 + lane) * 16];
            bfr.q[0] = bp[0];
            bfr.q[1] = bp[1];
            acc[nt] = __builtin_amdgcn_wmma_f32_16x16x32_bf16(
                false, a.v, false, bfr.v, (short)0, acc[nt], false, false);
        }
    }

#pragma unroll
    for (int nt = 0; nt < 4; ++nt)
#pragma unroll
        for (int r = 0; r < 8; ++r) {
            const int row = r + ((lane & 16) ? 8 : 0);
            z_s[wave][row][nt * 16 + (lane & 15)] = acc[nt][r];
        }
    __syncthreads();

    for (int idx = tid; idx < 16 * 64; idx += 128) {
        const int row = idx >> 6, col = idx & 63;
        const int b = bm0 + row, u = u0 + col;
        const float zi = z_s[0][row][col] + bias[0 * kU + u];
        const float zf = z_s[1][row][col] + bias[1 * kU + u];
        const float zg = z_s[2][row][col] + bias[2 * kU + u];
        const float zo = z_s[3][row][col] + bias[3 * kU + u];
        const float gi = sigmoidf_(zi), gf = sigmoidf_(zf), go = sigmoidf_(zo);
        const float gg = tanhf(zg);
        const float cp = cst[b * kU + u];
        const float ct = gf * cp + gi * gg;
        const float ht = go * tanhf(ct);
        const float kt = time_gate(times[b * kT + tstep], tau[u], shift[u]);
        const float cn = kt * ct + (1.0f - kt) * cp;
        const float hn = kt * ht + (1.0f - kt) * hprevf[b * kU + u];
        cst[b * kU + u]   = cn;
        hnewf[b * kU + u] = hn;
        hnewb[b * kU + u] = f32_to_bf16(hn);
    }
}

// ---------------- classifier head: logits + softmax (one wave per row) ------
__global__ __launch_bounds__(32) void head_kernel(
    const float* __restrict__ h1f, const float* __restrict__ wfc,
    const float* __restrict__ bfc, float* __restrict__ out, int tstep) {
    const int b = blockIdx.x;
    const int lane = threadIdx.x;
    float acc[kNC];
#pragma unroll
    for (int c = 0; c < kNC; ++c) acc[c] = 0.0f;
    for (int k = lane; k < kU; k += 32) {
        const float h = h1f[b * kU + k];
        const float* wr = wfc + k * kNC;
#pragma unroll
        for (int c = 0; c < kNC; ++c) acc[c] = fmaf(h, wr[c], acc[c]);
    }
#pragma unroll
    for (int c = 0; c < kNC; ++c) {
        for (int off = 16; off > 0; off >>= 1)
            acc[c] += __shfl_xor(acc[c], off, 32);
        acc[c] += bfc[c];
    }
    float m = acc[0];
#pragma unroll
    for (int c = 1; c < kNC; ++c) m = fmaxf(m, acc[c]);
    float s = 0.0f;
#pragma unroll
    for (int c = 0; c < kNC; ++c) {
        acc[c] = __expf(acc[c] - m);
        s += acc[c];
    }
    const float inv = 1.0f / s;
    if (lane < kNC) out[(b * kT + tstep) * kNC + lane] = acc[lane] * inv;
}

// ---------------- host launcher ----------------
extern "C" void kernel_launch(void* const* d_in, const int* in_sizes, int n_in,
                              void* d_out, int out_size, void* d_ws,
                              size_t ws_size, hipStream_t stream) {
    const float* inputs = (const float*)d_in[0];
    const float* times  = (const float*)d_in[1];
    const float* k0     = (const float*)d_in[2];
    const float* rk0    = (const float*)d_in[3];
    const float* b0     = (const float*)d_in[4];
    const float* tau0   = (const float*)d_in[5];
    const float* s0     = (const float*)d_in[6];
    const float* k1     = (const float*)d_in[7];
    const float* rk1    = (const float*)d_in[8];
    const float* b1     = (const float*)d_in[9];
    const float* tau1   = (const float*)d_in[10];
    const float* s1     = (const float*)d_in[11];
    const float* wfc    = (const float*)d_in[12];
    const float* bfc    = (const float*)d_in[13];
    float* out = (float*)d_out;

    // workspace carve-up (256B aligned)
    char* ws = (char*)d_ws;
    size_t off = 0;
    auto carve = [&](size_t bytes) -> void* {
        void* p = ws + off;
        off = (off + bytes + 255) & ~(size_t)255;
        return p;
    };
    unsigned short* w0p = (unsigned short*)carve((size_t)kKT0 * kNT * 512 * 2);
    unsigned short* w1p = (unsigned short*)carve((size_t)kKT1 * kNT * 512 * 2);

    const size_t hvN = (size_t)kB * kU;  // 65536 elements
    char* stateBase = (char*)carve(6 * hvN * 4 + 4 * hvN * 2);
    float* h0f[2] = {(float*)stateBase, (float*)stateBase + hvN};
    float* h1f[2] = {(float*)stateBase + 2 * hvN, (float*)stateBase + 3 * hvN};
    float* c0p = (float*)stateBase + 4 * hvN;
    float* c1p = (float*)stateBase + 5 * hvN;
    unsigned short* bb = (unsigned short*)((float*)stateBase + 6 * hvN);
    unsigned short* h0b[2] = {bb, bb + hvN};
    unsigned short* h1b[2] = {bb + 2 * hvN, bb + 3 * hvN};

    // prologue: zero state, pack weights to bf16 WMMA-B tiles
    const int stateUints = (int)((6 * hvN * 4 + 4 * hvN * 2) / 4);
    zero_state_kernel<<<2048, 256, 0, stream>>>((unsigned*)stateBase,
                                                stateUints);
    pack_weights_kernel<<<2048, 256, 0, stream>>>(k0, rk0, k1, rk1, w0p, w1p);

    // recurrence: 3 kernels per timestep, graph-capture friendly
    dim3 cellGrid(kB / 16, kU / 64);  // 8 x 8
    for (int t = 0; t < kT; ++t) {
        const int cur = t & 1, nxt = cur ^ 1;
        plstm_cell0<<<cellGrid, 128, 0, stream>>>(
            inputs, times, w0p, b0, tau0, s0, h0f[cur], h0b[cur], c0p,
            h0f[nxt], h0b[nxt], t);
        plstm_cell1<<<cellGrid, 128, 0, stream>>>(
            times, w1p, b1, tau1, s1, h0b[nxt], h1f[cur], h1b[cur], c1p,
            h1f[nxt], h1b[nxt], t);
        head_kernel<<<kB, 32, 0, stream>>>(h1f[nxt], wfc, bfc, out, t);
    }
    (void)in_sizes; (void)n_in; (void)out_size; (void)ws_size;
}
